// spiking_transformer_39367670235575
// MI455X (gfx1250) — compile-verified, matrix operand order checked by reading, not run on hardware
//
#include <hip/hip_runtime.h>

// ---------------------------------------------------------------------------
// Spiking transformer block (T=4, B=32, C=256, N=256, Hd=1024) for gfx1250.
// All GEMMs run on v_wmma_f32_16x16x32_bf16 (spikes are exactly {0,1} in bf16).
// ---------------------------------------------------------------------------

typedef __bf16 bf16;
typedef __attribute__((ext_vector_type(8)))  bf16  v8bf;
typedef __attribute__((ext_vector_type(16))) bf16  v16bf;
typedef __attribute__((ext_vector_type(8)))  float v8f;

union V16 { v16bf v; struct { v8bf lo, hi; } h; };

#define T_STEPS 4
#define TAU_INV 0.5f
#define BN_EPS  1e-5f

// ---------------------------------------------------------------------------
// f32 -> bf16 weight conversion (once per launch; deterministic)
// ---------------------------------------------------------------------------
__global__ void w_cvt_kernel(const float* __restrict__ in, bf16* __restrict__ out, int n) {
    int i = blockIdx.x * blockDim.x + threadIdx.x;
    if (i < n) out[i] = (bf16)in[i];
}

// ---------------------------------------------------------------------------
// LIF scan over T with transposed bf16 spike output.
//   X: [T, B, Cc, N] f32   (strideT = B*Cc*N)
//   S: [T, B, N, Cc] bf16  (same per-t element count)
// v += (x - v)/tau ; s = H(v - vth) ; v *= (1 - s)
// ---------------------------------------------------------------------------
__global__ void lif_tr_kernel(const float* __restrict__ X, bf16* __restrict__ S,
                              int Bd, int Cc, int N, float vth) {
    int tid = blockIdx.x * blockDim.x + threadIdx.x;
    int total = Bd * Cc * N;
    if (tid >= total) return;
    int c = tid % Cc;
    int n = (tid / Cc) % N;
    int b = tid / (Cc * N);
    size_t strideT = (size_t)Bd * Cc * N;
    size_t inBase  = ((size_t)b * Cc + c) * N + n;
    size_t outBase = ((size_t)b * N + n) * Cc + c;
    float v = 0.f;
#pragma unroll
    for (int t = 0; t < T_STEPS; ++t) {
        float x = X[inBase + (size_t)t * strideT];
        v += (x - v) * TAU_INV;
        float s = (v >= vth) ? 1.f : 0.f;
        v *= (1.f - s);
        S[outBase + (size_t)t * strideT] = (bf16)s;
    }
}

// ---------------------------------------------------------------------------
// Fused channel attention:
//   per (b,c): LIF(q), LIF(k), LIF(v) over t; attn_in[t] = sum_n sk*sv;
//   attn = LIF(attn_in, vth=0.5); y[t,n] = sq[t,n]*attn[t]  -> bf16 [t,b,n,c]
// One block per (b,c), 256 threads = one per token n.
// ---------------------------------------------------------------------------
__global__ __launch_bounds__(256)
void attn_fuse_kernel(const float* __restrict__ YQ, const float* __restrict__ YK,
                      const float* __restrict__ YV, bf16* __restrict__ QY,
                      int Bd, int C, int N) {
    int b = blockIdx.x / C;
    int c = blockIdx.x % C;
    int n = threadIdx.x;
    size_t strideT = (size_t)Bd * C * N;
    size_t base    = ((size_t)b * C + c) * N + n;

    float vq = 0.f, vk = 0.f, vv = 0.f;
    float sq[T_STEPS], prod[T_STEPS];
#pragma unroll
    for (int t = 0; t < T_STEPS; ++t) {
        size_t idx = base + (size_t)t * strideT;
        float q = YQ[idx];
        vq += (q - vq) * TAU_INV;
        float s = (vq >= 1.f) ? 1.f : 0.f; vq *= (1.f - s); sq[t] = s;
        float k = YK[idx];
        vk += (k - vk) * TAU_INV;
        float sk = (vk >= 1.f) ? 1.f : 0.f; vk *= (1.f - sk);
        float v = YV[idx];
        vv += (v - vv) * TAU_INV;
        float sv = (vv >= 1.f) ? 1.f : 0.f; vv *= (1.f - sv);
        prod[t] = sk * sv;
    }

    __shared__ float red[T_STEPS][256];
#pragma unroll
    for (int t = 0; t < T_STEPS; ++t) red[t][n] = prod[t];
    __syncthreads();
    for (int off = 128; off > 0; off >>= 1) {
        if (n < off) {
#pragma unroll
            for (int t = 0; t < T_STEPS; ++t) red[t][n] += red[t][n + off];
        }
        __syncthreads();
    }

    // attn LIF, vth = 0.5 (redundant per thread; tiny)
    float va = 0.f;
    size_t outBase = ((size_t)b * N + n) * C + c;
#pragma unroll
    for (int t = 0; t < T_STEPS; ++t) {
        float a = red[t][0];
        va += (a - va) * TAU_INV;
        float sa = (va >= 0.5f) ? 1.f : 0.f;
        va *= (1.f - sa);
        QY[outBase + (size_t)t * strideT] = (bf16)(sq[t] * sa);
    }
}

// ---------------------------------------------------------------------------
// WMMA bf16 GEMM with fused BN epilogue and optional bias/residual.
//   A  : [M, K]    bf16 weights, row-major (K contiguous)
//   Bm : [P, N, K] bf16 spikes, transposed per batch (K contiguous)
//   Out: [P, M, N] f32 ;  Out = (A@B + lin_bias)*bn_inv + bn_beta (+ Res)
// Block tile 128x128, 8 waves (2x4), each wave 64x32 via 4x2 WMMA tiles.
// ---------------------------------------------------------------------------
#define BM_T 128
#define BN_T 128

__global__ __launch_bounds__(256)
void gemm_bn_kernel(const bf16* __restrict__ A, const bf16* __restrict__ Bm,
                    const float* __restrict__ lin_bias,
                    const float* __restrict__ bng, const float* __restrict__ bnb,
                    const float* __restrict__ bnm, const float* __restrict__ bnv,
                    const float* __restrict__ Res, float* __restrict__ Out,
                    int M, int N, int K) {
    __shared__ float s_scale[BM_T];
    __shared__ float s_bias[BM_T];

    int mBlock = blockIdx.y * BM_T;
    int nBlock = blockIdx.x * BN_T;
    for (int i = threadIdx.x; i < BM_T; i += 256) {
        int ch = mBlock + i;
        float inv = bng[ch] * rsqrtf(bnv[ch] + BN_EPS);
        float bb  = bnb[ch] - bnm[ch] * inv;
        if (lin_bias) bb += lin_bias[ch] * inv;
        s_scale[i] = inv;
        s_bias[i]  = bb;
    }
    __syncthreads();

    int wave = threadIdx.x >> 5;
    int lane = threadIdx.x & 31;
    int wm = (wave >> 2) * 64;   // 0 or 64 within block tile
    int wn = (wave & 3) * 32;    // 0..96 within block tile
    int half = lane >> 4;        // lane half selects K sub-chunk
    int r    = lane & 15;        // row/col within 16

    const bf16* Ab = A + (size_t)mBlock * K;
    const bf16* Bb = Bm + (size_t)blockIdx.z * N * K + (size_t)nBlock * K;

    v8f acc[4][2];
#pragma unroll
    for (int i = 0; i < 4; ++i)
#pragma unroll
        for (int j = 0; j < 2; ++j) acc[i][j] = (v8f)(0.f);

    for (int k0 = 0; k0 < K; k0 += 32) {
        int kb = k0 + half * 8;
        if (k0 + 32 < K) {
            __builtin_prefetch(Ab + (size_t)(wm + r) * K + k0 + 32, 0, 1);
            __builtin_prefetch(Bb + (size_t)(wn + r) * K + k0 + 32, 0, 1);
        }
        v16bf afrag[4];
#pragma unroll
        for (int i = 0; i < 4; ++i) {
            const bf16* ap = Ab + (size_t)(wm + i * 16 + r) * K + kb;
            V16 u;
            u.h.lo = *(const v8bf*)(ap);
            u.h.hi = *(const v8bf*)(ap + 16);
            afrag[i] = u.v;
        }
        v16bf bfrag[2];
#pragma unroll
        for (int j = 0; j < 2; ++j) {
            const bf16* bp = Bb + (size_t)(wn + j * 16 + r) * K + kb;
            V16 u;
            u.h.lo = *(const v8bf*)(bp);
            u.h.hi = *(const v8bf*)(bp + 16);
            bfrag[j] = u.v;
        }
#pragma unroll
        for (int i = 0; i < 4; ++i)
#pragma unroll
            for (int j = 0; j < 2; ++j)
                acc[i][j] = __builtin_amdgcn_wmma_f32_16x16x32_bf16(
                    false, afrag[i], false, bfrag[j], (short)0, acc[i][j],
                    false, false);
    }

    // Epilogue: D fragment element e, lane -> (M = e + half*8, N = r)
#pragma unroll
    for (int i = 0; i < 4; ++i) {
#pragma unroll
        for (int j = 0; j < 2; ++j) {
#pragma unroll
            for (int e = 0; e < 8; ++e) {
                int mm = wm + i * 16 + e + half * 8;        // within block
                int nn = nBlock + wn + j * 16 + r;          // global N
                int gm = mBlock + mm;                       // global M
                float val = acc[i][j][e] * s_scale[mm] + s_bias[mm];
                size_t idx = (size_t)blockIdx.z * M * N + (size_t)gm * N + nn;
                if (Res) val += Res[idx];
                Out[idx] = val;
            }
        }
    }
}

// ---------------------------------------------------------------------------
// Host-side orchestration
// ---------------------------------------------------------------------------
extern "C" void kernel_launch(void* const* d_in, const int* in_sizes, int n_in,
                              void* d_out, int out_size, void* d_ws, size_t ws_size,
                              hipStream_t stream) {
    (void)in_sizes; (void)n_in; (void)out_size; (void)ws_size;

    const int T = 4, Bd = 32, C = 256, N = 256, Hd = 1024;
    const int P = T * Bd;  // 128 batched GEMMs

    const float* x     = (const float*)d_in[0];
    const float* wq    = (const float*)d_in[1];
    const float* wk    = (const float*)d_in[2];
    const float* wv    = (const float*)d_in[3];
    const float* wproj = (const float*)d_in[4];
    const float* bproj = (const float*)d_in[5];
    const float* qbn[4] = {(const float*)d_in[6],  (const float*)d_in[7],
                           (const float*)d_in[8],  (const float*)d_in[9]};
    const float* kbn[4] = {(const float*)d_in[10], (const float*)d_in[11],
                           (const float*)d_in[12], (const float*)d_in[13]};
    const float* vbn[4] = {(const float*)d_in[14], (const float*)d_in[15],
                           (const float*)d_in[16], (const float*)d_in[17]};
    const float* pbn[4] = {(const float*)d_in[18], (const float*)d_in[19],
                           (const float*)d_in[20], (const float*)d_in[21]};
    const float* w1  = (const float*)d_in[22];
    const float* b1  = (const float*)d_in[23];
    const float* bn1[4] = {(const float*)d_in[24], (const float*)d_in[25],
                           (const float*)d_in[26], (const float*)d_in[27]};
    const float* w2  = (const float*)d_in[28];
    const float* b2  = (const float*)d_in[29];
    const float* bn2[4] = {(const float*)d_in[30], (const float*)d_in[31],
                           (const float*)d_in[32], (const float*)d_in[33]};
    float* out = (float*)d_out;

    char* ws = (char*)d_ws;

    // --- bf16 weight region -------------------------------------------------
    bf16* wq_b    = (bf16*)(ws + 0);
    bf16* wk_b    = (bf16*)(ws + 131072);
    bf16* wv_b    = (bf16*)(ws + 262144);
    bf16* wproj_b = (bf16*)(ws + 393216);
    bf16* w1_b    = (bf16*)(ws + 524288);   // 1024*256
    bf16* w2_b    = (bf16*)(ws + 1048576);  // 256*1024
    char* arena   = ws + (2u << 20);

    // --- arena (with reuse) -------------------------------------------------
    const size_t SZ_SPIKE = (size_t)T * Bd * N * C * sizeof(bf16);  // 16.8 MB
    const size_t SZ_F32   = (size_t)T * Bd * C * N * sizeof(float); // 33.6 MB
    bf16*  XS0 = (bf16*)(arena + 0);
    float* YQ  = (float*)(arena + SZ_SPIKE);
    float* YK  = (float*)(arena + SZ_SPIKE + SZ_F32);
    float* YV  = (float*)(arena + SZ_SPIKE + 2 * SZ_F32);
    float* H   = (float*)(arena + 0);                 // reuses XS0+YQ/YK/YV (134 MB)
    bf16*  QY  = (bf16*)(arena + 134217728);          // 16.8 MB
    bf16*  XS1 = (bf16*)(arena + 134217728);          // reuses QY
    float* X1  = (float*)(arena + 150994944);         // 33.6 MB, live to the end
    bf16*  HS  = (bf16*)(arena + 184549376);          // 67.1 MB

    // 1) weights -> bf16
    w_cvt_kernel<<<(65536 + 255) / 256, 256, 0, stream>>>(wq, wq_b, 65536);
    w_cvt_kernel<<<(65536 + 255) / 256, 256, 0, stream>>>(wk, wk_b, 65536);
    w_cvt_kernel<<<(65536 + 255) / 256, 256, 0, stream>>>(wv, wv_b, 65536);
    w_cvt_kernel<<<(65536 + 255) / 256, 256, 0, stream>>>(wproj, wproj_b, 65536);
    w_cvt_kernel<<<(262144 + 255) / 256, 256, 0, stream>>>(w1, w1_b, 262144);
    w_cvt_kernel<<<(262144 + 255) / 256, 256, 0, stream>>>(w2, w2_b, 262144);

    // 2) input LIF -> transposed spikes
    {
        int total = Bd * C * N;
        lif_tr_kernel<<<(total + 255) / 256, 256, 0, stream>>>(x, XS0, Bd, C, N, 1.0f);
    }

    // 3) q/k/v GEMMs + BN (pre-LIF activations)
    {
        dim3 grid(N / BN_T, C / BM_T, P);
        gemm_bn_kernel<<<grid, 256, 0, stream>>>(wq_b, XS0, nullptr,
            qbn[0], qbn[1], qbn[2], qbn[3], nullptr, YQ, C, N, C);
        gemm_bn_kernel<<<grid, 256, 0, stream>>>(wk_b, XS0, nullptr,
            kbn[0], kbn[1], kbn[2], kbn[3], nullptr, YK, C, N, C);
        gemm_bn_kernel<<<grid, 256, 0, stream>>>(wv_b, XS0, nullptr,
            vbn[0], vbn[1], vbn[2], vbn[3], nullptr, YV, C, N, C);
    }

    // 4) fused channel attention -> qs*attn spikes (transposed bf16)
    attn_fuse_kernel<<<Bd * C, 256, 0, stream>>>(YQ, YK, YV, QY, Bd, C, N);

    // 5) proj GEMM + bproj + BN + residual x -> X1
    {
        dim3 grid(N / BN_T, C / BM_T, P);
        gemm_bn_kernel<<<grid, 256, 0, stream>>>(wproj_b, QY, bproj,
            pbn[0], pbn[1], pbn[2], pbn[3], x, X1, C, N, C);
    }

    // 6) MLP input LIF
    {
        int total = Bd * C * N;
        lif_tr_kernel<<<(total + 255) / 256, 256, 0, stream>>>(X1, XS1, Bd, C, N, 1.0f);
    }

    // 7) fc1 GEMM (1024x256x256) + b1 + BN1
    {
        dim3 grid(N / BN_T, Hd / BM_T, P);
        gemm_bn_kernel<<<grid, 256, 0, stream>>>(w1_b, XS1, b1,
            bn1[0], bn1[1], bn1[2], bn1[3], nullptr, H, Hd, N, C);
    }

    // 8) hidden LIF
    {
        int total = Bd * Hd * N;
        lif_tr_kernel<<<(total + 255) / 256, 256, 0, stream>>>(H, HS, Bd, Hd, N, 1.0f);
    }

    // 9) fc2 GEMM (256x1024x256) + b2 + BN2 + residual X1 -> out
    {
        dim3 grid(N / BN_T, C / BM_T, P);
        gemm_bn_kernel<<<grid, 256, 0, stream>>>(w2_b, HS, b2,
            bn2[0], bn2[1], bn2[2], bn2[3], X1, out, C, N, Hd);
    }
}
